// MultiHeadAttentionBlock_13511967113781
// MI455X (gfx1250) — compile-verified, matrix operand order checked by reading, not run on hardware
//
#include <hip/hip_runtime.h>
#include <hip/hip_fp16.h>
#include <stdint.h>

// ---------------------------------------------------------------------------
// MHA block for MI455X (gfx1250, wave32, WMMA, async global->LDS copies).
// B=2, S=2048, D=1024, H=16, DH=64. fp32 in/out, f16 WMMA internally.
// ---------------------------------------------------------------------------

typedef __attribute__((ext_vector_type(16))) _Float16 v16h;
typedef __attribute__((ext_vector_type(8)))  _Float16 v8h;
typedef __attribute__((ext_vector_type(4)))  _Float16 v4h;
typedef __attribute__((ext_vector_type(8)))  float    v8f;
typedef __attribute__((ext_vector_type(4)))  float    v4f;
typedef __attribute__((ext_vector_type(4)))  int      v4i;

#define B_  2
#define S_  2048
#define D_  1024
#define H_  16
#define DH_ 64
#define M_  (B_ * S_)   // 4096 total rows

// ---------------------------------------------------------------------------
// Async global -> LDS copy (16 bytes per lane), CDNA5 GLOBAL_LOAD_ASYNC_TO_LDS.
// ROCm 7.2 clang-22 declares the builtin with v4i* (int4 pointer) operands
// (probe: "cannot initialize a parameter of type 'int __vector(4) *'").
// Fall back to inline asm in GV addressing mode per the ISA if absent.
// ---------------------------------------------------------------------------
#if __has_builtin(__builtin_amdgcn_global_load_async_to_lds_b128)
#define ASYNC_COPY_BUILTIN 1
#endif
#if __has_builtin(__builtin_amdgcn_s_wait_asynccnt)
#define ASYNC_WAIT_BUILTIN 1
#endif

__device__ __forceinline__ void async_copy_b128(const _Float16* g, _Float16* l) {
#if defined(ASYNC_COPY_BUILTIN)
    __builtin_amdgcn_global_load_async_to_lds_b128(
        (v4i*)(uintptr_t)g, (v4i*)(uintptr_t)l, 0, 0);
#else
    unsigned           lds_off = (unsigned)(uintptr_t)l;
    unsigned long long gaddr   = (unsigned long long)(uintptr_t)g;
    asm volatile("global_load_async_to_lds_b128 %0, %1, off"
                 :: "v"(lds_off), "v"(gaddr) : "memory");
#endif
}

__device__ __forceinline__ void async_wait_all() {
#if defined(ASYNC_WAIT_BUILTIN)
    __builtin_amdgcn_s_wait_asynccnt(0);
#else
    asm volatile("s_wait_asynccnt 0x0" ::: "memory");
#endif
}

// ---------------------------------------------------------------------------
// WMMA helper: D = A(16x32 f16) * B(32x16 f16) + C(16x16 f32)
// ---------------------------------------------------------------------------
__device__ __forceinline__ v8f wmma16(v16h a, v16h b, v8f c) {
    return __builtin_amdgcn_wmma_f32_16x16x32_f16(
        /*neg_a=*/false, a, /*neg_b=*/false, b,
        /*c_mod=*/(short)0, c, /*reuse_a=*/false, /*reuse_b=*/false);
}

// ---------------------------------------------------------------------------
// Load a 16x32 f16 fragment in the documented CDNA5 A-matrix VGPR layout.
//   lanes 0-15 : M=lane,    halves 0-7 -> K=k0..k0+7,  halves 8-15 -> K=k0+16..k0+23
//   lanes 16-31: M=lane-16, halves 0-7 -> K=k0+8..+15, halves 8-15 -> K=k0+24..+31
// `base` may be LDS or global; `ld` is the row stride in halves (row stride
// bytes must be a multiple of 16 so both reads are b128).
// ---------------------------------------------------------------------------
__device__ __forceinline__ v16h load_frag(const _Float16* base, int row0,
                                          int k0, int ld) {
    const int lane = threadIdx.x & 31;
    const int r    = lane & 15;
    const int kk   = k0 + ((lane >> 4) << 3);   // +8 halves for upper half-wave
    const _Float16* p = base + (size_t)(row0 + r) * ld + kk;
    union { v16h v; v8h h[2]; } u;
    u.h[0] = *(const v8h*)(p);
    u.h[1] = *(const v8h*)(p + 16);
    return u.v;
}

// ---------------------------------------------------------------------------
// fp32 -> f16 conversion (vectorized x4). All array sizes are multiples of 4.
// ---------------------------------------------------------------------------
__global__ void __launch_bounds__(256)
cvt_f32_f16(const float* __restrict__ in, _Float16* __restrict__ out, int n4) {
    int i = blockIdx.x * blockDim.x + threadIdx.x;
    if (i < n4) {
        v4f x = *(const v4f*)(in + (size_t)i * 4);
        v4h y;
        y.x = (_Float16)x.x; y.y = (_Float16)x.y;
        y.z = (_Float16)x.z; y.w = (_Float16)x.w;
        *(v4h*)(out + (size_t)i * 4) = y;
    }
}

// ---------------------------------------------------------------------------
// GEMM: C[M,N] = A[M,K] @ W[N,K]^T   (y = x @ W.T, nn.Linear semantics)
// A, W f16 row-major; C f16 or f32. Block tile 128x128, 256 threads = 8 waves.
// Double-buffered LDS with async global->LDS prefetch of the next K-slice
// while WMMAs consume the current one. Safety: a wave's ds reads of the
// current buffer complete (s_wait_dscnt) before its WMMAs, which precede the
// end-of-iteration barrier; async writes are fenced by s_wait_asynccnt before
// the barrier signal.
// ---------------------------------------------------------------------------
template <bool OUT_HALF>
__global__ void __launch_bounds__(256)
gemm_a_wT(const _Float16* __restrict__ A, const _Float16* __restrict__ W,
          void* __restrict__ Cout, int Mtot, int N, int K) {
    __shared__ _Float16 sA[2][128][40];   // rows padded to 40 halves (80B, 16B-aligned)
    __shared__ _Float16 sB[2][128][40];

    const int bm   = blockIdx.x * 128;
    const int bn   = blockIdx.y * 128;
    const int tid  = threadIdx.x;
    const int wave = tid >> 5;
    const int wm   = (wave & 3) * 32;   // wave M offset in block tile
    const int wn   = (wave >> 2) * 64;  // wave N offset in block tile
    const int lane = tid & 31;
    const int mh   = (lane >> 4) * 8;   // C-layout row offset of upper half-wave
    const int nl   = lane & 15;         // C-layout column = lane % 16

    // Per-thread chunk coordinates for cooperative 128x32-half tile loads:
    // 512 b128 chunks per tile, 2 per thread.
    const int row0 = (tid * 2)     >> 2;
    const int col0 = ((tid * 2) & 3) * 8;
    const int row1 = (tid * 2 + 1) >> 2;
    const int col1 = ((tid * 2 + 1) & 3) * 8;

    auto load_tiles = [&](int k0, int buf) {
        async_copy_b128(A + (size_t)(bm + row0) * K + k0 + col0, &sA[buf][row0][col0]);
        async_copy_b128(W + (size_t)(bn + row0) * K + k0 + col0, &sB[buf][row0][col0]);
        async_copy_b128(A + (size_t)(bm + row1) * K + k0 + col1, &sA[buf][row1][col1]);
        async_copy_b128(W + (size_t)(bn + row1) * K + k0 + col1, &sB[buf][row1][col1]);
    };

    v8f acc[2][4];
#pragma unroll
    for (int i = 0; i < 2; i++)
#pragma unroll
        for (int j = 0; j < 4; j++) { v8f z = {}; acc[i][j] = z; }

    // Prologue: stage first K-slice.
    load_tiles(0, 0);
    async_wait_all();
    __syncthreads();

    for (int k0 = 0; k0 < K; k0 += 32) {
        const int cur = (k0 >> 5) & 1;
        // Prefetch next K-slice into the other buffer while computing.
        if (k0 + 32 < K) load_tiles(k0 + 32, cur ^ 1);

        v16h afr[2], bfr[4];
#pragma unroll
        for (int i = 0; i < 2; i++)
            afr[i] = load_frag(&sA[cur][0][0], wm + i * 16, 0, 40);
#pragma unroll
        for (int j = 0; j < 4; j++)
            bfr[j] = load_frag(&sB[cur][0][0], wn + j * 16, 0, 40);
#pragma unroll
        for (int i = 0; i < 2; i++)
#pragma unroll
            for (int j = 0; j < 4; j++)
                acc[i][j] = wmma16(afr[i], bfr[j], acc[i][j]);

        async_wait_all();
        __syncthreads();
    }

    // Epilogue: C-layout -> memory. VGPR r: lanes 0-15 row r, lanes 16-31 row r+8.
#pragma unroll
    for (int i = 0; i < 2; i++)
#pragma unroll
        for (int j = 0; j < 4; j++)
#pragma unroll
            for (int r = 0; r < 8; r++) {
                int m = bm + wm + i * 16 + mh + r;
                int n = bn + wn + j * 16 + nl;
                float val = acc[i][j][r];
                if (OUT_HALF)
                    ((_Float16*)Cout)[(size_t)m * N + n] = (_Float16)val;
                else
                    ((float*)Cout)[(size_t)m * N + n] = val;
            }
}

// ---------------------------------------------------------------------------
// Causal flash attention, one (b,h) per blockIdx.y, 128 query rows per block.
// 8 waves, each owns a 16-row query tile. Q fragments live in registers.
// K tile (128x64) streamed in with async global->LDS copies; V tile staged
// transposed (manual scatter — async copies cannot transpose). P restaged
// through a small per-wave LDS buffer (C-layout -> A-fragment layout) in
// 32-key chunks. Scale 1/sqrt(DH)=0.125 folded in; mask input unused (causal).
// ---------------------------------------------------------------------------
__global__ void __launch_bounds__(256)
flash_attn(const _Float16* __restrict__ Q, const _Float16* __restrict__ K,
           const _Float16* __restrict__ V, _Float16* __restrict__ O) {
    __shared__ _Float16 sK [128][72];     // [key][dh], 144B rows (16B-aligned)
    __shared__ _Float16 sVt[64][136];     // [dh][key], 272B rows (16B-aligned)
    __shared__ _Float16 sP [8][16][32];   // per-wave P chunk, 64B rows

    const int tid  = threadIdx.x;
    const int wave = tid >> 5;
    const int lane = tid & 31;
    const int mh   = (lane >> 4) * 8;
    const int nl   = lane & 15;

    const int qi      = blockIdx.x;
    const int bh      = blockIdx.y;
    const int b       = bh >> 4;          // H_ = 16
    const int h       = bh & 15;
    const int qbase   = qi * 128;
    const int rowbase = b * S_ + qbase;   // row in flattened [B*S, D]
    const int colbase = h * DH_;

    // Q fragments for this wave's 16 rows: DH=64 -> two K-chunks of 32.
    v16h aQ[2];
#pragma unroll
    for (int c = 0; c < 2; c++)
        aQ[c] = load_frag(Q + (size_t)rowbase * D_ + colbase,
                          wave * 16, c * 32, D_);

    float mrow[8], lrow[8];
#pragma unroll
    for (int r = 0; r < 8; r++) { mrow[r] = -1e30f; lrow[r] = 0.0f; }
    v8f o[4];
#pragma unroll
    for (int t = 0; t < 4; t++) { v8f z = {}; o[t] = z; }

    for (int j = 0; j <= qi; j++) {       // causal: only key blocks <= query block
        const int kb = j * 128;

        // Stage K tile (async, 1024 b128 chunks / 4 per thread) and V^T tile.
#pragma unroll
        for (int c = 0; c < 4; c++) {
            int chunk = tid * 4 + c;      // 0..1023
            int row   = chunk >> 3;       // 0..127 (key index)
            int col   = (chunk & 7) * 8;  // 0..56  (dh offset)
            const _Float16* gsrc =
                K + (size_t)(b * S_ + kb + row) * D_ + colbase + col;
            async_copy_b128(gsrc, &sK[row][col]);
            v8h vv =
                *(const v8h*)(V + (size_t)(b * S_ + kb + row) * D_ + colbase + col);
#pragma unroll
            for (int e = 0; e < 8; e++) sVt[col + e][row] = vv[e];
        }
        async_wait_all();
        __syncthreads();

        // Scores: S = Q @ K^T, 16x128 = 8 tiles, contraction DH=64 (2 WMMA steps).
        v8f s[8];
#pragma unroll
        for (int t = 0; t < 8; t++) {
            v8f z = {}; s[t] = z;
#pragma unroll
            for (int c = 0; c < 2; c++)
                s[t] = wmma16(aQ[c], load_frag(&sK[0][0], t * 16, c * 32, 72), s[t]);
        }

        // Online softmax per row (row r lives in 16 lanes of a half-wave).
#pragma unroll
        for (int r = 0; r < 8; r++) {
            const int qg = qbase + wave * 16 + mh + r;
            float pmax = -1e30f;
#pragma unroll
            for (int t = 0; t < 8; t++) {
                int   kg = kb + t * 16 + nl;
                float sv = s[t][r] * 0.125f;          // 1/sqrt(DH)
                sv       = (kg <= qg) ? sv : -1e30f;  // causal mask
                s[t][r]  = sv;
                pmax     = fmaxf(pmax, sv);
            }
#pragma unroll
            for (int off = 1; off < 16; off <<= 1)
                pmax = fmaxf(pmax, __shfl_xor(pmax, off, 32));
            float mnew     = fmaxf(mrow[r], pmax);
            float scaleOld = __expf(mrow[r] - mnew);
            mrow[r]        = mnew;
            float psum = 0.0f;
#pragma unroll
            for (int t = 0; t < 8; t++) {
                float p = __expf(s[t][r] - mnew);
                s[t][r] = p;
                psum += p;
            }
#pragma unroll
            for (int off = 1; off < 16; off <<= 1)
                psum += __shfl_xor(psum, off, 32);
            lrow[r] = lrow[r] * scaleOld + psum;
#pragma unroll
            for (int t = 0; t < 4; t++) o[t][r] *= scaleOld;
        }

        // O += P @ V in 4 chunks of 32 keys. Restage P through per-wave LDS to
        // convert C-layout -> A-fragment layout (wave-private: DScnt orders it).
#pragma unroll
        for (int c = 0; c < 4; c++) {
#pragma unroll
            for (int tt = 0; tt < 2; tt++) {
                int t = c * 2 + tt;
#pragma unroll
                for (int r = 0; r < 8; r++)
                    sP[wave][mh + r][tt * 16 + nl] = (_Float16)s[t][r];
            }
            v16h aP = load_frag(&sP[wave][0][0], 0, 0, 32);
#pragma unroll
            for (int t2 = 0; t2 < 4; t2++)
                o[t2] = wmma16(aP, load_frag(&sVt[0][0], t2 * 16, c * 32, 136), o[t2]);
        }
        __syncthreads();
    }

    // Normalize and store f16 attention output in [B,S,D] layout.
#pragma unroll
    for (int t = 0; t < 4; t++)
#pragma unroll
        for (int r = 0; r < 8; r++) {
            int sg = rowbase + wave * 16 + mh + r;
            int dd = colbase + t * 16 + nl;
            O[(size_t)sg * D_ + dd] = (_Float16)(o[t][r] / lrow[r]);
        }
}

// ---------------------------------------------------------------------------
// Host-side orchestration. Inputs: 0=query 1=key 2=value 3=mask(unused)
// 4=wq 5=wk 6=wv 7=wo. Workspace layout (f16): conversions + Q/K/V + attn out.
// ---------------------------------------------------------------------------
extern "C" void kernel_launch(void* const* d_in, const int* in_sizes, int n_in,
                              void* d_out, int out_size, void* d_ws, size_t ws_size,
                              hipStream_t stream) {
    const float* q  = (const float*)d_in[0];
    const float* k  = (const float*)d_in[1];
    const float* v  = (const float*)d_in[2];
    const float* wq = (const float*)d_in[4];
    const float* wk = (const float*)d_in[5];
    const float* wv = (const float*)d_in[6];
    const float* wo = (const float*)d_in[7];
    float* out = (float*)d_out;

    const size_t MD = (size_t)M_ * D_;   // 4 Mi elems
    const size_t DD = (size_t)D_ * D_;   // 1 Mi elems

    _Float16* ws  = (_Float16*)d_ws;
    _Float16* qh  = ws;                  // [M,D]  f16 inputs
    _Float16* kh  = qh  + MD;
    _Float16* vh  = kh  + MD;
    _Float16* wqh = vh  + MD;            // [D,D]  f16 weights
    _Float16* wkh = wqh + DD;
    _Float16* wvh = wkh + DD;
    _Float16* woh = wvh + DD;
    _Float16* Qp  = woh + DD;            // [M,D]  projected Q/K/V
    _Float16* Kp  = Qp  + MD;
    _Float16* Vp  = Kp  + MD;
    _Float16* Ao  = Vp  + MD;            // [M,D]  attention output

    const int CT = 256;
    auto cvt = [&](const float* src, _Float16* dst, size_t n) {
        int n4 = (int)(n / 4);
        cvt_f32_f16<<<(n4 + CT - 1) / CT, CT, 0, stream>>>(src, dst, n4);
    };
    cvt(q,  qh,  MD);
    cvt(k,  kh,  MD);
    cvt(v,  vh,  MD);
    cvt(wq, wqh, DD);
    cvt(wk, wkh, DD);
    cvt(wv, wvh, DD);
    cvt(wo, woh, DD);

    dim3 gg(M_ / 128, D_ / 128);         // 32 x 8 blocks
    gemm_a_wT<true><<<gg, 256, 0, stream>>>(qh, wqh, Qp, M_, D_, D_);
    gemm_a_wT<true><<<gg, 256, 0, stream>>>(kh, wkh, Kp, M_, D_, D_);
    gemm_a_wT<true><<<gg, 256, 0, stream>>>(vh, wvh, Vp, M_, D_, D_);

    flash_attn<<<dim3(S_ / 128, B_ * H_), 256, 0, stream>>>(Qp, Kp, Vp, Ao);

    gemm_a_wT<false><<<gg, 256, 0, stream>>>(Ao, woh, out, M_, D_, D_);
}
